// Block_53197464928933
// MI455X (gfx1250) — compile-verified, matrix operand order checked by reading, not run on hardware
//
#include <hip/hip_runtime.h>
#include <math.h>

// ---------------- problem constants ----------------
constexpr int Bb   = 32;
constexpr int Nn   = 577;
constexpr int D    = 768;
constexpr int Hh   = 12;
constexpr int Dh   = 64;
constexpr int HID  = 3072;
constexpr int TOK  = Bb * Nn;           // 18464
constexpr int QKVN = 3 * D;             // 2304

typedef __bf16 bf16_t;
typedef __bf16 v8bf  __attribute__((ext_vector_type(8)));
typedef __bf16 v16bf __attribute__((ext_vector_type(16)));
typedef float  v8f   __attribute__((ext_vector_type(8)));

// ---------------- CDNA5 async global->LDS path (guarded) ----------------
#if defined(__HIP_DEVICE_COMPILE__)
#if __has_builtin(__builtin_amdgcn_global_load_async_to_lds_b128) && \
    __has_builtin(__builtin_amdgcn_s_wait_asynccnt)
#define HAVE_ASYNC_LDS 1
#endif
#endif
#ifndef HAVE_ASYNC_LDS
#define HAVE_ASYNC_LDS 0
#endif

#if HAVE_ASYNC_LDS
typedef int v4i_t __attribute__((vector_size(16)));
typedef __attribute__((address_space(1))) v4i_t as1_v4i;
typedef __attribute__((address_space(3))) v4i_t as3_v4i;
static __device__ __forceinline__ void tile_cp16(const void* g, void* l) {
    // copy 16 bytes global -> LDS, tracked by ASYNCcnt (no VGPR data path)
    __builtin_amdgcn_global_load_async_to_lds_b128((as1_v4i*)g, (as3_v4i*)l, 0, 0);
}
static __device__ __forceinline__ void tile_wait() {
    __builtin_amdgcn_s_wait_asynccnt(0);
}
#else
static __device__ __forceinline__ void tile_cp16(const void* g, void* l) {
    *(uint4*)l = *(const uint4*)g;
}
static __device__ __forceinline__ void tile_wait() {}
#endif

static __device__ __forceinline__ v16bf cat16(v8bf lo, v8bf hi) {
    return __builtin_shufflevector(lo, hi, 0,1,2,3,4,5,6,7,8,9,10,11,12,13,14,15);
}
static __device__ __forceinline__ v8f wmma_bf16(v16bf a, v16bf b, v8f c) {
    // D = A(16x32) * B(32x16) + C, f32 accumulate
    return __builtin_amdgcn_wmma_f32_16x16x32_bf16(false, a, false, b, (short)0, c, false, false);
}
static __device__ __forceinline__ float gelu_exact(float v) {
    return 0.5f * v * (1.0f + erff(v * 0.70710678118654752f));
}

// ---------------- weight transpose + fp32 -> bf16 ----------------
// W [K x Nt] f32 (row-major)  ->  Wt [Nt x K] bf16 (row-major)
__global__ __launch_bounds__(256) void transpose_bf16_kernel(
    const float* __restrict__ W, bf16_t* __restrict__ Wt, int K, int Nt) {
    __shared__ float tile[32][33];
    const int tx = threadIdx.x & 31;
    const int ty = threadIdx.x >> 5;          // 0..7
    const int n0 = blockIdx.x * 32;           // over Nt
    const int k0 = blockIdx.y * 32;           // over K
#pragma unroll
    for (int j = ty; j < 32; j += 8) {
        int k = k0 + j, n = n0 + tx;
        tile[j][tx] = (k < K && n < Nt) ? W[(size_t)k * Nt + n] : 0.0f;
    }
    __syncthreads();
#pragma unroll
    for (int j = ty; j < 32; j += 8) {
        int n = n0 + j, k = k0 + tx;
        if (n < Nt && k < K) Wt[(size_t)n * K + k] = (bf16_t)tile[tx][j];
    }
}

// ---------------- LayerNorm (f32 in) -> bf16 out ----------------
// one wave (32 lanes) per token row of 768
__global__ __launch_bounds__(256) void ln_bf16_kernel(
    const float* __restrict__ x, const float* __restrict__ g,
    const float* __restrict__ bta, bf16_t* __restrict__ out, int rows) {
    const int wid  = (int)((blockIdx.x * blockDim.x + threadIdx.x) >> 5);
    const int lane = threadIdx.x & 31;
    if (wid >= rows) return;
    const float* px = x + (size_t)wid * D;
    float s = 0.0f, s2 = 0.0f;
#pragma unroll
    for (int i = lane; i < D; i += 32) { float v = px[i]; s += v; s2 += v * v; }
#pragma unroll
    for (int off = 16; off > 0; off >>= 1) {
        s  += __shfl_xor(s,  off, 32);
        s2 += __shfl_xor(s2, off, 32);
    }
    const float mu  = s * (1.0f / D);
    const float var = s2 * (1.0f / D) - mu * mu;
    const float rs  = rsqrtf(var + 1e-5f);
    bf16_t* po = out + (size_t)wid * D;
#pragma unroll
    for (int i = lane; i < D; i += 32)
        po[i] = (bf16_t)((px[i] - mu) * rs * g[i] + bta[i]);
}

// ---------------- tiled bf16 GEMM with WMMA ----------------
// C[M x Ntot] = A[M x K] * B[K x Ntot], with B given pre-transposed: Bt[Ntot x K].
// Block tile 128x128, K-step 32, double-buffered LDS fed by async global->LDS copies.
// 8 waves: wm = w&1 (2 x 64 rows), wn = w>>1 (4 x 32 cols).
enum { EPI_QKV = 0, EPI_PROJ = 1, EPI_FC1 = 2, EPI_FC2 = 3 };

template <int EPI>
__global__ __launch_bounds__(256) void gemm_bf16_kernel(
    const bf16_t* __restrict__ A, const bf16_t* __restrict__ Bt,
    int M, int Ntot, int K,
    const float* __restrict__ bias, const float* __restrict__ resid,
    float* __restrict__ Cf, bf16_t* __restrict__ Cb,
    bf16_t* __restrict__ Qb, bf16_t* __restrict__ Kb, bf16_t* __restrict__ Vb) {
    __shared__ __align__(16) bf16_t As[2][128 * 32];
    __shared__ __align__(16) bf16_t Bs[2][128 * 32];

    const int tid  = threadIdx.x;
    const int lane = tid & 31;
    const int w    = tid >> 5;
    const int wm   = w & 1;
    const int wn   = w >> 1;
    const int r16  = lane & 15;
    const int kh   = (lane >> 4) << 3;      // 0 or 8 (k offset within 32-k tile)
    const int blockM = blockIdx.y * 128;
    const int blockN = blockIdx.x * 128;

    // global->LDS mover geometry: each thread moves 16 bf16 (32B) of one tile row
    const int ldr = tid >> 1;               // 0..127
    const int ldc = (tid & 1) << 4;         // 0 or 16 (bf16 elems)
    const int aRow = blockM + ldr;
    const int bRow = blockN + ldr;
    const bool aOk = aRow < M;
    const bool bOk = bRow < Ntot;
    // OOB rows leave stale LDS data; it only feeds C rows/cols never stored.

    const v8f vz = {0.f,0.f,0.f,0.f,0.f,0.f,0.f,0.f};
    v8f acc[4][2];
#pragma unroll
    for (int mf = 0; mf < 4; ++mf)
#pragma unroll
        for (int nf = 0; nf < 2; ++nf) acc[mf][nf] = vz;

    // prologue: tile k0=0 into buffer 0
    if (aOk) {
        tile_cp16(A + (size_t)aRow * K + ldc,     &As[0][ldr * 32 + ldc]);
        tile_cp16(A + (size_t)aRow * K + ldc + 8, &As[0][ldr * 32 + ldc + 8]);
    }
    if (bOk) {
        tile_cp16(Bt + (size_t)bRow * K + ldc,     &Bs[0][ldr * 32 + ldc]);
        tile_cp16(Bt + (size_t)bRow * K + ldc + 8, &Bs[0][ldr * 32 + ldc + 8]);
    }
    tile_wait();
    __syncthreads();

    int cur = 0;
    for (int k0 = 0; k0 < K; k0 += 32) {
        // issue async copy of the next tile into the other buffer (overlaps WMMAs)
        const int kn = k0 + 32;
        if (kn < K) {
            const int nb = cur ^ 1;
            if (aOk) {
                tile_cp16(A + (size_t)aRow * K + kn + ldc,     &As[nb][ldr * 32 + ldc]);
                tile_cp16(A + (size_t)aRow * K + kn + ldc + 8, &As[nb][ldr * 32 + ldc + 8]);
            }
            if (bOk) {
                tile_cp16(Bt + (size_t)bRow * K + kn + ldc,     &Bs[nb][ldr * 32 + ldc]);
                tile_cp16(Bt + (size_t)bRow * K + kn + ldc + 8, &Bs[nb][ldr * 32 + ldc + 8]);
            }
        }

        v16bf af[4], bfr[2];
#pragma unroll
        for (int mf = 0; mf < 4; ++mf) {
            const int row = wm * 64 + mf * 16 + r16;
            v8bf lo = *(const v8bf*)(&As[cur][row * 32 + kh]);
            v8bf hi = *(const v8bf*)(&As[cur][row * 32 + 16 + kh]);
            af[mf] = cat16(lo, hi);
        }
#pragma unroll
        for (int nf = 0; nf < 2; ++nf) {
            const int row = wn * 32 + nf * 16 + r16;
            v8bf lo = *(const v8bf*)(&Bs[cur][row * 32 + kh]);
            v8bf hi = *(const v8bf*)(&Bs[cur][row * 32 + 16 + kh]);
            bfr[nf] = cat16(lo, hi);
        }
#pragma unroll
        for (int mf = 0; mf < 4; ++mf)
#pragma unroll
            for (int nf = 0; nf < 2; ++nf)
                acc[mf][nf] = wmma_bf16(af[mf], bfr[nf], acc[mf][nf]);

        tile_wait();          // our async writes to the next buffer are done
        __syncthreads();      // everyone finished reading cur / writing next
        cur ^= 1;
    }

    // epilogue: element (m,n) for vgpr r: m_local = r + 8*(lane>>4), n_local = lane&15
#pragma unroll
    for (int mf = 0; mf < 4; ++mf)
#pragma unroll
        for (int nf = 0; nf < 2; ++nf)
#pragma unroll
            for (int r = 0; r < 8; ++r) {
                const int m = blockM + wm * 64 + mf * 16 + ((lane >> 4) << 3) + r;
                const int n = blockN + wn * 32 + nf * 16 + (lane & 15);
                if (m >= M || n >= Ntot) continue;
                float v = acc[mf][nf][r];
                if constexpr (EPI == EPI_QKV) {
                    const int sect = n / D;            // 0=q 1=k 2=v
                    const int d    = n - sect * D;
                    const int hh   = d >> 6;
                    const int dh   = d & 63;
                    const int bbi  = m / Nn;
                    const int tk   = m - bbi * Nn;
                    const size_t o = ((size_t)(bbi * Hh + hh) * Nn + tk) * Dh + dh;
                    const bf16_t bv = (bf16_t)v;
                    if (sect == 0)      Qb[o] = bv;
                    else if (sect == 1) Kb[o] = bv;
                    else                Vb[o] = bv;
                } else if constexpr (EPI == EPI_PROJ) {
                    v += bias[n] + resid[(size_t)m * Ntot + n];
                    Cf[(size_t)m * Ntot + n] = v;
                } else if constexpr (EPI == EPI_FC1) {
                    v = gelu_exact(v + bias[n]);
                    Cb[(size_t)m * Ntot + n] = (bf16_t)v;
                } else { // EPI_FC2
                    v += bias[n] + resid[(size_t)m * Ntot + n];
                    Cf[(size_t)m * Ntot + n] = v;
                }
            }
}

// ---------------- flash attention (bf16 WMMA, f32 online softmax) ----------------
// grid: B*H*ceil(N/64) blocks of 128 threads (4 waves, 16 query rows each)
constexpr int MT = (Nn + 63) / 64;   // 10 row tiles

__global__ __launch_bounds__(128) void attn_kernel(
    const bf16_t* __restrict__ Qb, const bf16_t* __restrict__ Kb,
    const bf16_t* __restrict__ Vb, const float* __restrict__ cpmask,
    bf16_t* __restrict__ Ob) {
    __shared__ __align__(16) bf16_t Ks[64 * 64];        // [key][d]
    __shared__ __align__(16) bf16_t Vt[64 * 64];        // [d][key] (transposed)
    __shared__ __align__(16) bf16_t Ps[4][16 * 64];     // per-wave P staging

    const int bid = blockIdx.x;
    const int mt  = bid % MT;
    const int bh  = bid / MT;
    const int b   = bh / Hh;
    const int h   = bh - b * Hh;
    const size_t base = (size_t)bh * Nn * Dh;

    const int tid  = threadIdx.x;
    const int lane = tid & 31;
    const int w    = tid >> 5;                  // 0..3
    const int r16  = lane & 15;
    const int kh   = (lane >> 4) << 3;          // 0 or 8
    const int m0   = mt * 64 + w * 16;          // wave's first query row

    // Q fragments (A-layout), clamp OOB rows (results discarded at store)
    const int qrow  = m0 + r16;
    const int qrowc = qrow < Nn ? qrow : (Nn - 1);
    v16bf qf[2];
#pragma unroll
    for (int ks = 0; ks < 2; ++ks) {
        v8bf lo = *(const v8bf*)(Qb + base + (size_t)qrowc * Dh + ks * 32 + kh);
        v8bf hi = *(const v8bf*)(Qb + base + (size_t)qrowc * Dh + ks * 32 + 16 + kh);
        qf[ks] = cat16(lo, hi);
    }

    const v8f vz = {0.f,0.f,0.f,0.f,0.f,0.f,0.f,0.f};
    v8f o[4] = {vz, vz, vz, vz};
    float rm[8], rl[8];
#pragma unroll
    for (int r = 0; r < 8; ++r) { rm[r] = -3.0e38f; rl[r] = 0.0f; }

    const int kr = tid >> 1;                    // loader: key row 0..63
    const int kc = (tid & 1) << 5;              // 0 or 32 (d offset)

    for (int j0 = 0; j0 < Nn; j0 += 64) {
        __syncthreads();                        // protect previous tile reads
        const int krow = j0 + kr;
        if (krow < Nn) {
            // K tile: async global->LDS (no VGPR staging)
            tile_cp16(Kb + base + (size_t)krow * Dh + kc,     Ks + kr * 64 + kc);
            tile_cp16(Kb + base + (size_t)krow * Dh + kc + 8, Ks + kr * 64 + kc + 8);
            // V tile: transpose into [d][key] (element stores)
            const bf16_t* pv = Vb + base + (size_t)krow * Dh + kc;
#pragma unroll
            for (int d2 = 0; d2 < 32; ++d2) Vt[(kc + d2) * 64 + kr] = pv[d2];
        } else {
            const uint4 z{};
            *(uint4*)(Ks + kr * 64 + kc)     = z;
            *(uint4*)(Ks + kr * 64 + kc + 8) = z;
#pragma unroll
            for (int d2 = 0; d2 < 32; ++d2) Vt[(kc + d2) * 64 + kr] = (bf16_t)0.0f;
        }
        tile_wait();
        __syncthreads();

        // S = Q * K^T  (16 x 64)
        v8f s[4] = {vz, vz, vz, vz};
#pragma unroll
        for (int ks = 0; ks < 2; ++ks)
#pragma unroll
            for (int nf = 0; nf < 4; ++nf) {
                const int row = nf * 16 + r16;  // key row for this lane's column
                v8bf lo = *(const v8bf*)(Ks + row * 64 + ks * 32 + kh);
                v8bf hi = *(const v8bf*)(Ks + row * 64 + ks * 32 + 16 + kh);
                s[nf] = wmma_bf16(qf[ks], cat16(lo, hi), s[nf]);
            }

        // scale * cp_mask, OOB keys -> -inf
        const float sc = 0.125f;                // 64^-0.5
#pragma unroll
        for (int nf = 0; nf < 4; ++nf)
#pragma unroll
            for (int r = 0; r < 8; ++r) {
                const int mq  = m0 + ((lane >> 4) << 3) + r;
                const int mqc = mq < Nn ? mq : (Nn - 1);
                const int nk  = j0 + nf * 16 + r16;
                float val;
                if (nk < Nn) val = s[nf][r] * sc * cpmask[(size_t)mqc * Nn + nk];
                else         val = -3.0e38f;
                s[nf][r] = val;
            }

        // online softmax per row r (row lives on 16 lanes of same half-wave)
#pragma unroll
        for (int r = 0; r < 8; ++r) {
            float mx = s[0][r];
#pragma unroll
            for (int nf = 1; nf < 4; ++nf) mx = fmaxf(mx, s[nf][r]);
#pragma unroll
            for (int off = 8; off > 0; off >>= 1) mx = fmaxf(mx, __shfl_xor(mx, off, 16));
            const float mnew  = fmaxf(rm[r], mx);
            const float alpha = __expf(rm[r] - mnew);
            rm[r] = mnew;
#pragma unroll
            for (int df = 0; df < 4; ++df) o[df][r] *= alpha;
            float psum = 0.0f;
#pragma unroll
            for (int nf = 0; nf < 4; ++nf) {
                const float p = __expf(s[nf][r] - mnew);
                s[nf][r] = p;
                psum += p;
            }
#pragma unroll
            for (int off = 8; off > 0; off >>= 1) psum += __shfl_xor(psum, off, 16);
            rl[r] = rl[r] * alpha + psum;
        }

        // stage P into per-wave LDS to re-enter A-layout
#pragma unroll
        for (int nf = 0; nf < 4; ++nf)
#pragma unroll
            for (int r = 0; r < 8; ++r) {
                const int ml = ((lane >> 4) << 3) + r;
                Ps[w][ml * 64 + nf * 16 + r16] = (bf16_t)s[nf][r];
            }

        // O += P * V   (P: 16x64, V^T staged as [d][key])
#pragma unroll
        for (int ks = 0; ks < 2; ++ks) {
            v8bf plo = *(const v8bf*)(&Ps[w][r16 * 64 + ks * 32 + kh]);
            v8bf phi = *(const v8bf*)(&Ps[w][r16 * 64 + ks * 32 + 16 + kh]);
            const v16bf pa = cat16(plo, phi);
#pragma unroll
            for (int df = 0; df < 4; ++df) {
                const int row = df * 16 + r16;  // dh column for this lane
                v8bf vlo = *(const v8bf*)(Vt + row * 64 + ks * 32 + kh);
                v8bf vhi = *(const v8bf*)(Vt + row * 64 + ks * 32 + 16 + kh);
                o[df] = wmma_bf16(pa, cat16(vlo, vhi), o[df]);
            }
        }
    }

    // normalize + store to [b, n, h*64 + dh] bf16
#pragma unroll
    for (int df = 0; df < 4; ++df)
#pragma unroll
        for (int r = 0; r < 8; ++r) {
            const int mq = m0 + ((lane >> 4) << 3) + r;
            if (mq < Nn) {
                const float v = o[df][r] / rl[r];
                Ob[((size_t)(b * Nn + mq)) * D + h * Dh + df * 16 + r16] = (bf16_t)v;
            }
        }
}

// ---------------- host orchestration ----------------
extern "C" void kernel_launch(void* const* d_in, const int* in_sizes, int n_in,
                              void* d_out, int out_size, void* d_ws, size_t ws_size,
                              hipStream_t stream) {
    (void)in_sizes; (void)n_in; (void)out_size; (void)ws_size;
    const float* x     = (const float*)d_in[0];
    const float* cp    = (const float*)d_in[1];
    const float* ln1g  = (const float*)d_in[2];
    const float* ln1b  = (const float*)d_in[3];
    const float* wqkv  = (const float*)d_in[4];
    const float* wproj = (const float*)d_in[5];
    const float* bproj = (const float*)d_in[6];
    const float* ln2g  = (const float*)d_in[7];
    const float* ln2b  = (const float*)d_in[8];
    const float* wfc1  = (const float*)d_in[9];
    const float* bfc1  = (const float*)d_in[10];
    const float* wfc2  = (const float*)d_in[11];
    const float* bfc2  = (const float*)d_in[12];
    float* out = (float*)d_out;

    // workspace layout (bytes, 256-aligned blocks)
    char* ws = (char*)d_ws;
    bf16_t* wqkvT = (bf16_t*)(ws + 0);                       //  3,538,944
    bf16_t* wprojT = (bf16_t*)(ws + 3538944);                //  1,179,648
    bf16_t* wfc1T  = (bf16_t*)(ws + 4718592);                //  4,718,592
    bf16_t* wfc2T  = (bf16_t*)(ws + 9437184);                //  4,718,592
    bf16_t* lnbuf  = (bf16_t*)(ws + 14155776);               // 28,360,704 (ln out / attn out)
    float*  x1     = (float*) (ws + 42516480);               // 56,721,408
    bf16_t* qbuf   = (bf16_t*)(ws + 99237888);               // qkv (85MB) then h1 (113MB)
    bf16_t* kbuf   = qbuf + (size_t)Bb * Hh * Nn * Dh;
    bf16_t* vbuf   = kbuf + (size_t)Bb * Hh * Nn * Dh;
    bf16_t* h1     = qbuf;                                   // reuse after attention

    // 1) weight transpose + convert
    transpose_bf16_kernel<<<dim3(QKVN / 32, D / 32), 256, 0, stream>>>(wqkv,  wqkvT, D,   QKVN);
    transpose_bf16_kernel<<<dim3(D / 32,   D / 32), 256, 0, stream>>>(wproj, wprojT, D,   D);
    transpose_bf16_kernel<<<dim3(HID / 32, D / 32), 256, 0, stream>>>(wfc1,  wfc1T,  D,   HID);
    transpose_bf16_kernel<<<dim3(D / 32, HID / 32), 256, 0, stream>>>(wfc2,  wfc2T,  HID, D);

    // 2) LN1 -> bf16
    ln_bf16_kernel<<<TOK / 8, 256, 0, stream>>>(x, ln1g, ln1b, lnbuf, TOK);

    // 3) QKV GEMM (scatter to [B,H,N,Dh] bf16)
    gemm_bf16_kernel<EPI_QKV><<<dim3(QKVN / 128, (TOK + 127) / 128), 256, 0, stream>>>(
        lnbuf, wqkvT, TOK, QKVN, D, nullptr, nullptr, nullptr, nullptr, qbuf, kbuf, vbuf);

    // 4) attention (writes bf16 [TOK, D] into lnbuf)
    attn_kernel<<<Bb * Hh * MT, 128, 0, stream>>>(qbuf, kbuf, vbuf, cp, lnbuf);

    // 5) proj GEMM + bias + residual(x) -> x1 (f32)
    gemm_bf16_kernel<EPI_PROJ><<<dim3(D / 128, (TOK + 127) / 128), 256, 0, stream>>>(
        lnbuf, wprojT, TOK, D, D, bproj, x, x1, nullptr, nullptr, nullptr, nullptr);

    // 6) LN2 -> bf16
    ln_bf16_kernel<<<TOK / 8, 256, 0, stream>>>(x1, ln2g, ln2b, lnbuf, TOK);

    // 7) FC1 GEMM + bias + GELU -> h1 (bf16)
    gemm_bf16_kernel<EPI_FC1><<<dim3(HID / 128, (TOK + 127) / 128), 256, 0, stream>>>(
        lnbuf, wfc1T, TOK, HID, D, bfc1, nullptr, nullptr, h1, nullptr, nullptr, nullptr);

    // 8) FC2 GEMM + bias + residual(x1) -> d_out (f32)
    gemm_bf16_kernel<EPI_FC2><<<dim3(D / 128, (TOK + 127) / 128), 256, 0, stream>>>(
        h1, wfc2T, TOK, D, HID, bfc2, x1, out, nullptr, nullptr, nullptr, nullptr);
}
